// FasterRcnn_64372969832531
// MI455X (gfx1250) — matrix operand order repaired and checked by reading, not compiled, not run on hardware
//
#include <hip/hip_runtime.h>

// ---------------------------------------------------------------------------
// Faster R-CNN forward for MI455X (gfx1250, wave32, WMMA).
// All heavy GEMMs (conv1..3 via im2col, rpn conv, fc1, fc2) use
// v_wmma_f32_16x16x32_bf16 with fp32 accumulation. Each wave computes a
// 16x64 output tile (1 A fragment reused across 4 B fragments -> 4 WMMAs
// per K-step) with unguarded, fully unrolled global loads.
// ---------------------------------------------------------------------------

typedef __attribute__((ext_vector_type(16))) __bf16 v16bf;
typedef __attribute__((ext_vector_type(8)))  __bf16 v8bf;
typedef __attribute__((ext_vector_type(8)))  float  v8f;

#define IMG 512
#define FH 16
#define TOPN 1000
#define NANC 2304            // 16*16*9
#define BBOX_CLAMP 4.135166556742356f   // log(1000/16)

// ---------------------------------------------------------------------------
// Wave-level WMMA GEMM:  C[M,N] = relu(A[Mp,K] * Bt[N,K]^T + bias)
// A row-major bf16 with Mp = ceil(M/16)*16 physical rows (pad rows may hold
// garbage; they only affect output rows >= M which are never stored).
// Bt is B transposed (N x K row-major) bf16. K % 32 == 0, N % 64 == 0.
// One wave per 16x64 tile: grid = (N/64, ceil(M/16)).
// ---------------------------------------------------------------------------
__global__ __launch_bounds__(32) void k_gemm_bf16_wmma(
    const __bf16* __restrict__ A, const __bf16* __restrict__ Bt,
    const float* __restrict__ bias, float* __restrict__ C,
    __bf16* __restrict__ Cbf, int M, int N, int K, int relu)
{
    const int tn   = blockIdx.x * 64;
    const int tm   = blockIdx.y * 16;
    const int lane = threadIdx.x;
    const int half = lane >> 4;      // 0: lanes 0-15, 1: lanes 16-31
    const int lm   = lane & 15;

    // A fragment: lanes 0-15 carry K[0..15]; lanes 16-31 K[16..31]
    const __bf16* aptr = A + (size_t)(tm + lm) * K + half * 16;
    // B fragments: column (tn+16*t+lm), contiguous 16 K values per half-wave
    const __bf16* bptr[4];
    #pragma unroll
    for (int t = 0; t < 4; ++t)
        bptr[t] = Bt + (size_t)(tn + 16 * t + lm) * K + half * 16;

    v8f acc[4] = {{}, {}, {}, {}};
    for (int k0 = 0; k0 < K; k0 += 32) {
        v8bf lo = *(const v8bf*)(aptr + k0);
        v8bf hi = *(const v8bf*)(aptr + k0 + 8);
        v16bf afrag = __builtin_shufflevector(
            lo, hi, 0,1,2,3,4,5,6,7,8,9,10,11,12,13,14,15);
        v16bf bfrag[4];
        #pragma unroll
        for (int t = 0; t < 4; ++t)
            bfrag[t] = *(const v16bf*)(bptr[t] + k0);

        if (k0 + 64 < K) {                       // -> global_prefetch_b8
            __builtin_prefetch(aptr + k0 + 64, 0, 1);
            #pragma unroll
            for (int t = 0; t < 4; ++t)
                __builtin_prefetch(bptr[t] + k0 + 64, 0, 1);
        }
        #pragma unroll
        for (int t = 0; t < 4; ++t)
            acc[t] = __builtin_amdgcn_wmma_f32_16x16x32_bf16(
                         false, afrag, false, bfrag[t], (short)0, acc[t],
                         false, false);
    }

    #pragma unroll
    for (int t = 0; t < 4; ++t) {
        const int   n  = tn + 16 * t + lm;
        const float bv = bias ? bias[n] : 0.0f;
        #pragma unroll
        for (int v = 0; v < 8; ++v) {
            int row = tm + half * 8 + v;         // C/D layout: VGPR v, half -> M+8
            if (row < M) {
                float x = acc[t][v] + bv;
                if (relu) x = fmaxf(x, 0.0f);
                C[(size_t)row * N + n] = x;
                if (Cbf) Cbf[(size_t)row * N + n] = (__bf16)x;
            }
        }
    }
}

// ---------------------------------------------------------------------------
// Stem: normalize + 7x7 s2 p3 conv + relu.  img CHW 3x512x512 -> out NHWC 256x256x64
// ---------------------------------------------------------------------------
__global__ __launch_bounds__(256) void k_conv0(
    const float* __restrict__ img, const float* __restrict__ w,
    const float* __restrict__ bias, float* __restrict__ out)
{
    int idx = blockIdx.x * 256 + threadIdx.x;          // (y*256+x)*64+o
    if (idx >= 256 * 256 * 64) return;
    int o = idx & 63, x = (idx >> 6) & 255, y = idx >> 14;
    const float mean[3] = {0.485f, 0.456f, 0.406f};
    const float stdv[3] = {0.229f, 0.224f, 0.225f};
    float s = bias[o];
    for (int c = 0; c < 3; ++c) {
        float inv = 1.0f / stdv[c];
        for (int r = 0; r < 7; ++r) {
            int iy = y * 2 - 3 + r;
            if (iy < 0 || iy >= IMG) continue;
            for (int q = 0; q < 7; ++q) {
                int ix = x * 2 - 3 + q;
                if (ix < 0 || ix >= IMG) continue;
                float v = (img[(c * IMG + iy) * IMG + ix] - mean[c]) * inv;
                s += v * w[((o * 3 + c) * 7 + r) * 7 + q];
            }
        }
    }
    out[idx] = fmaxf(s, 0.0f);
}

// maxpool 3x3 s2 p1, NHWC 256x256x64 -> 128x128x64
__global__ __launch_bounds__(256) void k_maxpool(
    const float* __restrict__ in, float* __restrict__ out)
{
    int idx = blockIdx.x * 256 + threadIdx.x;
    if (idx >= 128 * 128 * 64) return;
    int o = idx & 63, x = (idx >> 6) & 127, y = idx >> 13;
    float m = -3.4e38f;
    for (int r = 0; r < 3; ++r) {
        int iy = y * 2 - 1 + r;
        if (iy < 0 || iy >= 256) continue;
        for (int q = 0; q < 3; ++q) {
            int ix = x * 2 - 1 + q;
            if (ix < 0 || ix >= 256) continue;
            m = fmaxf(m, in[(iy * 256 + ix) * 64 + o]);
        }
    }
    out[idx] = m;
}

// im2col (3x3, pad 1) NHWC fp32 -> bf16 A[M=Hout*Wout][K=Cin*9], k = i*9+r*3+s
__global__ __launch_bounds__(256) void k_im2col(
    const float* __restrict__ in, __bf16* __restrict__ A,
    int Hin, int Win, int Cin, int Hout, int Wout, int stride)
{
    int K = Cin * 9;
    int tid = blockIdx.x * 256 + threadIdx.x;
    if (tid >= Hout * Wout * K) return;
    int m = tid / K, k = tid - m * K;
    int i = k / 9, rs = k - i * 9, r = rs / 3, s = rs - r * 3;
    int oy = m / Wout, ox = m - oy * Wout;
    int iy = oy * stride - 1 + r, ix = ox * stride - 1 + s;
    float v = 0.0f;
    if (iy >= 0 && iy < Hin && ix >= 0 && ix < Win)
        v = in[((size_t)iy * Win + ix) * Cin + i];
    A[tid] = (__bf16)v;
}

// fp32 -> bf16 (conv weights are already [N][K] row-major == Bt)
__global__ __launch_bounds__(256) void k_cvt(
    const float* __restrict__ src, __bf16* __restrict__ dst, int n)
{
    int i = blockIdx.x * 256 + threadIdx.x;
    if (i < n) dst[i] = (__bf16)src[i];
}

// fp32 [K][N] -> bf16 [N][K] (fc weights)
__global__ __launch_bounds__(256) void k_cvt_tr(
    const float* __restrict__ src, __bf16* __restrict__ dst, int K, int N)
{
    int i = blockIdx.x * 256 + threadIdx.x;
    if (i >= K * N) return;
    int k = i / N, n = i - k * N;
    dst[(size_t)n * K + k] = (__bf16)src[i];
}

// RPN 1x1 heads over t NHWC [256][512]: obj [m*9+a], bdf [(m*9+a)*4+comp]
__global__ __launch_bounds__(256) void k_rpn_heads(
    const float* __restrict__ t, const float* __restrict__ cw,
    const float* __restrict__ cb, const float* __restrict__ bw,
    const float* __restrict__ bb, float* __restrict__ obj, float* __restrict__ bdf)
{
    int tid = blockIdx.x * 256 + threadIdx.x;
    if (tid >= 256 * 45) return;
    int m = tid / 45, ch = tid - m * 45;
    const float* trow = t + (size_t)m * 512;
    if (ch < 9) {
        float s = cb[ch];
        const float* wr = cw + (size_t)ch * 512;
        for (int k = 0; k < 512; ++k) s += trow[k] * wr[k];
        obj[m * 9 + ch] = s;
    } else {
        int j = ch - 9;
        float s = bb[j];
        const float* wr = bw + (size_t)j * 512;
        for (int k = 0; k < 512; ++k) s += trow[k] * wr[k];
        int a = j >> 2, comp = j & 3;
        bdf[(m * 9 + a) * 4 + comp] = s;
    }
}

// anchors (generated analytically) + proposal decode (weights 1,1,1,1)
__global__ __launch_bounds__(256) void k_props(
    const float* __restrict__ bdf, float* __restrict__ props)
{
    int i = blockIdx.x * 256 + threadIdx.x;
    if (i >= NANC) return;
    int cell = i / 9, a = i - cell * 9;
    int ratio = a / 3, sidx = a - ratio * 3;
    const float scales[3] = {128.0f, 256.0f, 512.0f};
    const float ratios[3] = {0.5f, 1.0f, 2.0f};
    float hr = sqrtf(ratios[ratio]);
    float ws = (1.0f / hr) * scales[sidx];
    float hs = hr * scales[sidx];
    float cx0 = rintf(-ws * 0.5f), cy0 = rintf(-hs * 0.5f);
    float cx1 = rintf( ws * 0.5f), cy1 = rintf( hs * 0.5f);
    int cy = cell / FH, cx = cell - cy * FH;
    float ax1 = cx * 32.0f + cx0, ay1 = cy * 32.0f + cy0;
    float ax2 = cx * 32.0f + cx1, ay2 = cy * 32.0f + cy1;
    float w = ax2 - ax1, h = ay2 - ay1;
    float ctx = ax1 + 0.5f * w, cty = ay1 + 0.5f * h;
    float dx = bdf[i * 4 + 0], dy = bdf[i * 4 + 1];
    float dw = fminf(bdf[i * 4 + 2], BBOX_CLAMP);
    float dh = fminf(bdf[i * 4 + 3], BBOX_CLAMP);
    float pcx = dx * w + ctx, pcy = dy * h + cty;
    float pw = expf(dw) * w,  ph = expf(dh) * h;
    props[i * 4 + 0] = pcx - 0.5f * pw;
    props[i * 4 + 1] = pcy - 0.5f * ph;
    props[i * 4 + 2] = pcx + 0.5f * pw;
    props[i * 4 + 3] = pcy + 0.5f * ph;
}

// deterministic top-1000 by rank (ties broken by index, matches lax.top_k)
__global__ __launch_bounds__(256) void k_topk(
    const float* __restrict__ scores, int* __restrict__ top_idx)
{
    __shared__ float sh[NANC];
    for (int j = threadIdx.x; j < NANC; j += 256) sh[j] = scores[j];
    __syncthreads();
    int i = blockIdx.x * 256 + threadIdx.x;
    if (i >= NANC) return;
    float si = sh[i];
    int rank = 0;
    for (int j = 0; j < NANC; ++j) {
        float sj = sh[j];
        rank += (sj > si) || (sj == si && j < i);
    }
    if (rank < TOPN) top_idx[rank] = i;
}

__global__ __launch_bounds__(256) void k_rois(
    const float* __restrict__ props, const int* __restrict__ top_idx,
    float* __restrict__ rois)
{
    int r = blockIdx.x * 256 + threadIdx.x;
    if (r >= TOPN) return;
    int i = top_idx[r];
    for (int j = 0; j < 4; ++j)
        rois[r * 4 + j] = fminf(fmaxf(props[i * 4 + j], 0.0f), (float)IMG);
}

// ROI-align 14x14 sampled -> 7x7 pooled; writes bf16 row [c*49 + py*7 + px]
__global__ __launch_bounds__(256) void k_roi_align(
    const float* __restrict__ feat, const float* __restrict__ rois,
    __bf16* __restrict__ Ar)
{
    const int r = blockIdx.x;
    const int c = threadIdx.x;                         // channels c and c+256
    const float sc = 1.0f / 32.0f;
    float x1 = rois[r * 4 + 0] * sc, y1 = rois[r * 4 + 1] * sc;
    float x2 = rois[r * 4 + 2] * sc, y2 = rois[r * 4 + 3] * sc;
    float rw = fmaxf(x2 - x1, 1.0f), rh = fmaxf(y2 - y1, 1.0f);
    __bf16* out = Ar + (size_t)r * 25088;
    for (int py = 0; py < 7; ++py)
        for (int px = 0; px < 7; ++px) {
            float a0 = 0.0f, a1 = 0.0f;
            for (int gy = 0; gy < 2; ++gy)
                for (int gx = 0; gx < 2; ++gx) {
                    float gxf = ((float)(2 * px + gx) + 0.5f) * (1.0f / 14.0f);
                    float gyf = ((float)(2 * py + gy) + 0.5f) * (1.0f / 14.0f);
                    float sx = fminf(fmaxf(x1 + gxf * rw, 0.0f), 15.0f);
                    float sy = fminf(fmaxf(y1 + gyf * rh, 0.0f), 15.0f);
                    int x0 = (int)floorf(sx), y0 = (int)floorf(sy);
                    int x1i = min(x0 + 1, 15), y1i = min(y0 + 1, 15);
                    float lx = sx - (float)x0, ly = sy - (float)y0;
                    float hx = 1.0f - lx, hy = 1.0f - ly;
                    const float* f00 = feat + ((size_t)y0  * 16 + x0 ) * 512;
                    const float* f01 = feat + ((size_t)y0  * 16 + x1i) * 512;
                    const float* f10 = feat + ((size_t)y1i * 16 + x0 ) * 512;
                    const float* f11 = feat + ((size_t)y1i * 16 + x1i) * 512;
                    float w00 = hy * hx, w01 = hy * lx, w10 = ly * hx, w11 = ly * lx;
                    a0 += w00 * f00[c]     + w01 * f01[c]     + w10 * f10[c]     + w11 * f11[c];
                    a1 += w00 * f00[c+256] + w01 * f01[c+256] + w10 * f10[c+256] + w11 * f11[c+256];
                }
            out[(size_t)c * 49 + py * 7 + px]         = (__bf16)(a0 * 0.25f);
            out[(size_t)(c + 256) * 49 + py * 7 + px] = (__bf16)(a1 * 0.25f);
        }
}

// cls/bbox heads + softmax + class-box decode + clip + pack output [1000][5][5]
__global__ __launch_bounds__(256) void k_final(
    const float* __restrict__ h2, const float* __restrict__ cw,
    const float* __restrict__ cb, const float* __restrict__ bw,
    const float* __restrict__ bb, const float* __restrict__ rois,
    float* __restrict__ out)
{
    int r = blockIdx.x * 256 + threadIdx.x;
    if (r >= TOPN) return;
    float logits[6], breg[24];
    for (int o = 0; o < 6; ++o)  logits[o] = cb[o];
    for (int o = 0; o < 24; ++o) breg[o]   = bb[o];
    const float* hr = h2 + (size_t)r * 1024;
    for (int k = 0; k < 1024; ++k) {
        float h = hr[k];
        const float* cwr = cw + (size_t)k * 6;
        const float* bwr = bw + (size_t)k * 24;
        for (int o = 0; o < 6; ++o)  logits[o] += h * cwr[o];
        for (int o = 0; o < 24; ++o) breg[o]   += h * bwr[o];
    }
    float mx = logits[0];
    for (int o = 1; o < 6; ++o) mx = fmaxf(mx, logits[o]);
    float ex[6], se = 0.0f;
    for (int o = 0; o < 6; ++o) { ex[o] = expf(logits[o] - mx); se += ex[o]; }
    float bx1 = rois[r*4+0], by1 = rois[r*4+1], bx2 = rois[r*4+2], by2 = rois[r*4+3];
    float w = bx2 - bx1, h = by2 - by1;
    float ctx = bx1 + 0.5f * w, cty = by1 + 0.5f * h;
    for (int c = 1; c < 6; ++c) {
        float dx = breg[c*4+0] * 0.1f, dy = breg[c*4+1] * 0.1f;
        float dw = fminf(breg[c*4+2] * 0.2f, BBOX_CLAMP);
        float dh = fminf(breg[c*4+3] * 0.2f, BBOX_CLAMP);
        float pcx = dx * w + ctx, pcy = dy * h + cty;
        float pw = expf(dw) * w,  ph = expf(dh) * h;
        float ox1 = fminf(fmaxf(pcx - 0.5f * pw, 0.0f), (float)IMG);
        float oy1 = fminf(fmaxf(pcy - 0.5f * ph, 0.0f), (float)IMG);
        float ox2 = fminf(fmaxf(pcx + 0.5f * pw, 0.0f), (float)IMG);
        float oy2 = fminf(fmaxf(pcy + 0.5f * ph, 0.0f), (float)IMG);
        float* orow = out + ((size_t)r * 5 + (c - 1)) * 5;
        orow[0] = ox1; orow[1] = oy1; orow[2] = ox2; orow[3] = oy2;
        orow[4] = ex[c] / se;
    }
}

// ---------------------------------------------------------------------------
static inline size_t au(size_t x) { return (x + 255) & ~(size_t)255; }

extern "C" void kernel_launch(void* const* d_in, const int* in_sizes, int n_in,
                              void* d_out, int out_size, void* d_ws, size_t ws_size,
                              hipStream_t stream) {
    (void)in_sizes; (void)n_in; (void)out_size; (void)ws_size;
    const float* images    = (const float*)d_in[0];
    const float* c0_w      = (const float*)d_in[1];
    const float* c0_b      = (const float*)d_in[2];
    const float* c1_w      = (const float*)d_in[3];
    const float* c1_b      = (const float*)d_in[4];
    const float* c2_w      = (const float*)d_in[5];
    const float* c2_b      = (const float*)d_in[6];
    const float* c3_w      = (const float*)d_in[7];
    const float* c3_b      = (const float*)d_in[8];
    const float* rpnc_w    = (const float*)d_in[9];
    const float* rpnc_b    = (const float*)d_in[10];
    const float* rpncls_w  = (const float*)d_in[11];
    const float* rpncls_b  = (const float*)d_in[12];
    const float* rpnbb_w   = (const float*)d_in[13];
    const float* rpnbb_b   = (const float*)d_in[14];
    const float* fc1_w     = (const float*)d_in[15];
    const float* fc1_b     = (const float*)d_in[16];
    const float* fc2_w     = (const float*)d_in[17];
    const float* fc2_b     = (const float*)d_in[18];
    const float* cls_w     = (const float*)d_in[19];
    const float* cls_b     = (const float*)d_in[20];
    const float* bbox_w    = (const float*)d_in[21];
    const float* bbox_b    = (const float*)d_in[22];
    float* out = (float*)d_out;

    // workspace carve-up (~143 MB total)
    char* p = (char*)d_ws; size_t off = 0;
    #define CARVE(ptr, type, bytes) type* ptr = (type*)(p + off); off += au(bytes);
    CARVE(conv0_out, float, 256u*256u*64u*4u)        // 16.8 MB NHWC
    CARVE(pool_out,  float, 128u*128u*64u*4u)        //  4.2 MB NHWC
    CARVE(c1_out,    float, 64u*64u*128u*4u)
    CARVE(c2_out,    float, 32u*32u*256u*4u)
    CARVE(feat,      float, 16u*16u*512u*4u)
    CARVE(trpn,      float, 16u*16u*512u*4u)
    CARVE(obj,       float, NANC*4u)
    CARVE(bdf,       float, NANC*4u*4u)
    CARVE(props,     float, NANC*4u*4u)
    CARVE(tidx,      int,   TOPN*4u)
    CARVE(rois,      float, TOPN*4u*4u)
    CARVE(h1,        float, (size_t)TOPN*1024u*4u)
    CARVE(h1b,       __bf16,(size_t)1008u*1024u*2u)  // 1008 = ceil(1000/16)*16 rows
    CARVE(h2,        float, (size_t)TOPN*1024u*4u)
    CARVE(acol,      __bf16,(size_t)4096u*576u*2u)   // im2col buf (max: c1)
    CARVE(wb,        __bf16,(size_t)25088u*1024u*2u) // bf16 weights (max: fc1)
    CARVE(Ar,        __bf16,(size_t)1008u*25088u*2u) // ROI features (padded rows)
    #undef CARVE

    // --- stem ---
    k_conv0  <<<(256*256*64)/256, 256, 0, stream>>>(images, c0_w, c0_b, conv0_out);
    k_maxpool<<<(128*128*64)/256, 256, 0, stream>>>(conv0_out, pool_out);

    // --- conv GEMMs via im2col + WMMA (K = Cin*9, multiple of 32; N % 64 == 0) ---
    // c1: M=4096 N=128 K=576
    k_im2col<<<(4096*576)/256, 256, 0, stream>>>(pool_out, acol, 128,128,64, 64,64, 2);
    k_cvt   <<<(128*576+255)/256, 256, 0, stream>>>(c1_w, wb, 128*576);
    k_gemm_bf16_wmma<<<dim3(128/64, 4096/16), 32, 0, stream>>>(acol, wb, c1_b, c1_out, nullptr, 4096,128,576, 1);
    // c2: M=1024 N=256 K=1152
    k_im2col<<<(1024*1152)/256, 256, 0, stream>>>(c1_out, acol, 64,64,128, 32,32, 2);
    k_cvt   <<<(256*1152+255)/256, 256, 0, stream>>>(c2_w, wb, 256*1152);
    k_gemm_bf16_wmma<<<dim3(256/64, 1024/16), 32, 0, stream>>>(acol, wb, c2_b, c2_out, nullptr, 1024,256,1152, 1);
    // c3: M=256 N=512 K=2304 -> feat (NHWC 16x16x512)
    k_im2col<<<(256*2304)/256, 256, 0, stream>>>(c2_out, acol, 32,32,256, 16,16, 2);
    k_cvt   <<<(512*2304+255)/256, 256, 0, stream>>>(c3_w, wb, 512*2304);
    k_gemm_bf16_wmma<<<dim3(512/64, 256/16), 32, 0, stream>>>(acol, wb, c3_b, feat, nullptr, 256,512,2304, 1);
    // rpn conv: M=256 N=512 K=4608 (stride 1)
    k_im2col<<<(256*4608)/256, 256, 0, stream>>>(feat, acol, 16,16,512, 16,16, 1);
    k_cvt   <<<(512*4608+255)/256, 256, 0, stream>>>(rpnc_w, wb, 512*4608);
    k_gemm_bf16_wmma<<<dim3(512/64, 256/16), 32, 0, stream>>>(acol, wb, rpnc_b, trpn, nullptr, 256,512,4608, 1);

    // --- RPN heads, proposals, top-k, ROI gather ---
    k_rpn_heads<<<(256*45+255)/256, 256, 0, stream>>>(trpn, rpncls_w, rpncls_b, rpnbb_w, rpnbb_b, obj, bdf);
    k_props<<<NANC/256, 256, 0, stream>>>(bdf, props);
    k_topk <<<NANC/256, 256, 0, stream>>>(obj, tidx);
    k_rois <<<(TOPN+255)/256, 256, 0, stream>>>(props, tidx, rois);

    // --- ROI align -> bf16 feature rows [1000 x 25088] (buffer padded to 1008) ---
    k_roi_align<<<TOPN, 256, 0, stream>>>(feat, rois, Ar);

    // --- fc1: M=1000 N=1024 K=25088 (dominant GEMM, ~51 GFLOP) ---
    k_cvt_tr<<<(25088*1024+255)/256, 256, 0, stream>>>(fc1_w, wb, 25088, 1024);
    k_gemm_bf16_wmma<<<dim3(1024/64, (TOPN+15)/16), 32, 0, stream>>>(Ar, wb, fc1_b, h1, h1b, TOPN,1024,25088, 1);
    // --- fc2: M=1000 N=1024 K=1024 ---
    k_cvt_tr<<<(1024*1024+255)/256, 256, 0, stream>>>(fc2_w, wb, 1024, 1024);
    k_gemm_bf16_wmma<<<dim3(1024/64, (TOPN+15)/16), 32, 0, stream>>>(h1b, wb, fc2_b, h2, nullptr, TOPN,1024,1024, 1);

    // --- heads + softmax + decode + output pack ---
    k_final<<<(TOPN+255)/256, 256, 0, stream>>>(h2, cls_w, cls_b, bbox_w, bbox_b, rois, out);
}
